// Sparse2L_16741782520301
// MI455X (gfx1250) — compile-verified, matrix operand order checked by reading, not run on hardware
//
#include <hip/hip_runtime.h>

typedef unsigned short u16;
typedef unsigned int   u32;
typedef __attribute__((ext_vector_type(4)))  unsigned int u32x4;
typedef __attribute__((ext_vector_type(4)))  int          v4i;
typedef __attribute__((ext_vector_type(8)))  float        v8f;
typedef __attribute__((ext_vector_type(16))) __bf16       v16bf;

#define TILE_M 128
#define TILE_N 128
#define KC     64
#define LDSTR  72   // 64 k-elems + 8 pad; 144B rows keep 16B alignment

#if defined(__gfx1250__) && __has_builtin(__builtin_amdgcn_global_load_async_to_lds_b128)
#define HAVE_ASYNC 1
#else
#define HAVE_ASYNC 0
#endif

__device__ __forceinline__ u16 f2bf(float x) {
  u32 u = __float_as_uint(x);
  u32 r = u + 0x7FFFu + ((u >> 16) & 1u);   // round-to-nearest-even
  return (u16)(r >> 16);
}

__device__ __forceinline__ void await_async() {
#if defined(__gfx1250__)
#if __has_builtin(__builtin_amdgcn_s_wait_asynccnt)
  __builtin_amdgcn_s_wait_asynccnt(0);
#else
  asm volatile("s_wait_asynccnt 0" ::: "memory");
#endif
#endif
}

#if HAVE_ASYNC
__device__ __forceinline__ void async_cp16(const u16* g, u16* l) {
  __builtin_amdgcn_global_load_async_to_lds_b128(
      (__attribute__((address_space(1))) v4i*)(u16*)g,
      (__attribute__((address_space(3))) v4i*)l, 0, 0);
}
#endif

// Build a 16x32 (A) or 32x16 (B) bf16 WMMA fragment from an LDS tile stored
// k-contiguous. ISA layout: lane 0-15 row/col = lane%16; VGPRs 0-3 hold
// k = 8*half + 0..7, VGPRs 4-7 hold k = 16 + 8*half + 0..7.
__device__ __forceinline__ v16bf frag_ld(const u16* s, int row, int ks, int hl) {
  union { v16bf v; u32x4 q[2]; } f;
  const u16* base = s + row * LDSTR + ks * 32 + 8 * hl;
  f.q[0] = *(const u32x4*)(base);        // k = 8*hl .. 8*hl+7
  f.q[1] = *(const u32x4*)(base + 16);   // k = 16+8*hl .. 16+8*hl+7
  return f.v;
}

// C = A(bf16, [M,K] row-major) x Bn(bf16, [N,K] n-major, k-contiguous).
// 128x128 block / 256 threads (8 waves of 64x32), f32 accumulation,
// KC=64 double-buffered LDS, one barrier per chunk.
template <typename Epi>
__device__ __forceinline__ void gemm_run(const u16* __restrict__ Ag,
                                         const u16* __restrict__ Bg,
                                         int K, Epi epi) {
  __shared__ __align__(16) u16 As[2][TILE_M * LDSTR];
  __shared__ __align__(16) u16 Bs[2][TILE_N * LDSTR];

  const int m0   = blockIdx.x * TILE_M;
  const int n0   = blockIdx.y * TILE_N;
  const int tid  = threadIdx.x;
  const int lane = tid & 31;
  const int w    = tid >> 5;
  const int wm   = w >> 2;      // 0..1  (64 rows each)
  const int wn   = w & 3;       // 0..3  (32 cols each)
  const int hl   = lane >> 4;
  const int lr   = lane & 15;

  v8f acc[4][2];
#pragma unroll
  for (int i = 0; i < 4; ++i)
#pragma unroll
    for (int j = 0; j < 2; ++j)
      acc[i][j] = v8f{};

  const int row = tid >> 1;              // 0..127
  const int kp  = (tid & 1) * 32;        // 0 / 32
  const size_t gAoff = (size_t)(m0 + row) * K + kp;
  const size_t gBoff = (size_t)(n0 + row) * K + kp;
  const int nch = K / KC;

  auto compute = [&](int buf) {
#pragma unroll
    for (int ks = 0; ks < 2; ++ks) {
      const u16* sA = As[buf];
      const u16* sB = Bs[buf];
      v16bf bfr[2];
#pragma unroll
      for (int j = 0; j < 2; ++j)
        bfr[j] = frag_ld(sB, wn * 32 + j * 16 + lr, ks, hl);
#pragma unroll
      for (int i = 0; i < 4; ++i) {
        v16bf af = frag_ld(sA, wm * 64 + i * 16 + lr, ks, hl);
#pragma unroll
        for (int j = 0; j < 2; ++j)
          acc[i][j] = __builtin_amdgcn_wmma_f32_16x16x32_bf16(
              false, af, false, bfr[j], (short)0, acc[i][j], false, false);
      }
    }
  };

#if HAVE_ASYNC
  auto stage = [&](int buf, int kk) {
    const u16* ga = Ag + gAoff + kk;
    const u16* gb = Bg + gBoff + kk;
    u16* la = &As[buf][row * LDSTR + kp];
    u16* lb = &Bs[buf][row * LDSTR + kp];
#pragma unroll
    for (int q = 0; q < 4; ++q) {
      async_cp16(ga + q * 8, la + q * 8);
      async_cp16(gb + q * 8, lb + q * 8);
    }
  };
  stage(0, 0);
  await_async();
  __syncthreads();
  for (int kc = 0; kc < nch; ++kc) {
    const int cur = kc & 1;
    if (kc + 1 < nch) stage(1 - cur, (kc + 1) * KC);
    compute(cur);
    await_async();
    __syncthreads();
  }
#else
  u32x4 ra[4], rb[4];
  auto fetch = [&](int kk) {
    const u32x4* ga = (const u32x4*)(Ag + gAoff + kk);
    const u32x4* gb = (const u32x4*)(Bg + gBoff + kk);
#pragma unroll
    for (int q = 0; q < 4; ++q) { ra[q] = ga[q]; rb[q] = gb[q]; }
  };
  auto commit = [&](int buf) {
    u16* la = &As[buf][row * LDSTR + kp];
    u16* lb = &Bs[buf][row * LDSTR + kp];
#pragma unroll
    for (int q = 0; q < 4; ++q) {
      *(u32x4*)(la + q * 8) = ra[q];
      *(u32x4*)(lb + q * 8) = rb[q];
    }
  };
  fetch(0);
  commit(0);
  __syncthreads();
  for (int kc = 0; kc < nch; ++kc) {
    const int cur = kc & 1;
    if (kc + 1 < nch) fetch((kc + 1) * KC);
    compute(cur);
    if (kc + 1 < nch) commit(1 - cur);
    __syncthreads();
  }
#endif

  // D layout: VGPR r holds m = r + 8*(lane/16); n = lane%16
#pragma unroll
  for (int i = 0; i < 4; ++i)
#pragma unroll
    for (int j = 0; j < 2; ++j) {
      const int gn = n0 + wn * 32 + j * 16 + lr;
#pragma unroll
      for (int r = 0; r < 8; ++r) {
        const int gm = m0 + wm * 64 + i * 16 + r + 8 * hl;
        epi(gm, gn, acc[i][j][r]);
      }
    }
}

// ---- GEMM kernels with fused epilogues -----------------------------------

// E = exp(-(u @ V)),  A = ubf [4096,256], B = VTbf [1024,256]
__global__ void __launch_bounds__(256)
k_gemm_uV(const u16* __restrict__ A, const u16* __restrict__ B,
          float* __restrict__ E) {
  gemm_run(A, B, 256, [&](int m, int n, float v) {
    E[(size_t)m * 1024 + n] = __expf(-v);
  });
}

// outputs = x @ W; mode 0: Rbf = bf16(outputs - inputs); mode 1: d_out = outputs
__global__ void __launch_bounds__(256)
k_gemm_xW(const u16* __restrict__ A, const u16* __restrict__ B,
          const float* __restrict__ inp, u16* __restrict__ Rbf,
          float* __restrict__ outF, int mode) {
  gemm_run(A, B, 4096, [&](int m, int n, float v) {
    size_t idx = (size_t)m * 512 + n;
    if (mode == 0) Rbf[idx] = f2bf(v - inp[idx]);
    else           outF[idx] = v;
  });
}

// grad_x GEMM + soft-threshold x update. A = Rbf [4096,512], B = Wbf [4096,512]
__global__ void __launch_bounds__(256)
k_gemm_gradx(const u16* __restrict__ A, const u16* __restrict__ B,
             const float* __restrict__ E, float* __restrict__ x,
             u16* __restrict__ xbf) {
  gemm_run(A, B, 512, [&](int m, int p, float g) {
    int pm = p >> 6, pc = p & 63;
    int j  = (pm >> 1) * 32 + (pc >> 1);                  // pooled index
    float t  = 5e-5f * (1.0f + E[(size_t)m * 1024 + j]);  // LR*GAMMA*0.5*(1+E)
    size_t idx = (size_t)m * 4096 + p;
    float xn = x[idx] - 0.002f * g;                       // x - LR*2*(R@W^T)
    float y  = fmaxf(xn - t, 0.0f) + fminf(xn + t, 0.0f);
    x[idx]   = y;
    xbf[idx] = f2bf(y);
  });
}

// grad_u GEMM + soft-threshold u update. A = Mbf [4096,1024], B = Vbf [256,1024]
__global__ void __launch_bounds__(256)
k_gemm_gradu(const u16* __restrict__ A, const u16* __restrict__ B,
             float* __restrict__ u, u16* __restrict__ ubf) {
  gemm_run(A, B, 1024, [&](int m, int n, float g) {
    size_t idx = (size_t)m * 256 + n;
    float un = u[idx] - 0.001f * g;                       // u - LR*grad_u
    float y  = fmaxf(un - 1e-5f, 0.0f) + fminf(un + 1e-5f, 0.0f);
    u[idx]   = y;
    ubf[idx] = f2bf(y);
  });
}

// ---- elementwise helpers --------------------------------------------------

__global__ void k_fill_f32(float* p, float v, int n) {
  int i = blockIdx.x * 256 + threadIdx.x;
  if (i < n) p[i] = v;
}
__global__ void k_fill_u16(u16* p, u16 v, int n) {
  int i = blockIdx.x * 256 + threadIdx.x;
  if (i < n) p[i] = v;
}
__global__ void k_cvt_bf(const float* __restrict__ s, u16* __restrict__ d, int n) {
  int i = blockIdx.x * 256 + threadIdx.x;
  if (i < n) d[i] = f2bf(s[i]);
}
// d[c*rows + r] = bf16(s[r*cols + c])
__global__ void k_tr_cvt(const float* __restrict__ s, u16* __restrict__ d,
                         int rows, int cols) {
  int i = blockIdx.x * 256 + threadIdx.x;
  if (i < rows * cols) {
    int r = i / cols, c = i % cols;
    d[(size_t)c * rows + r] = f2bf(s[i]);
  }
}
__global__ void k_negR(const float* __restrict__ in, u16* __restrict__ R, int n) {
  int i = blockIdx.x * 256 + threadIdx.x;
  if (i < n) R[i] = f2bf(-in[i]);
}
// 2x2 max-pool of the 64x64 code map + Mbf = bf16(-0.05 * E * x_pool)
__global__ void k_pool(const float* __restrict__ x, const float* __restrict__ E,
                       u16* __restrict__ M) {
  int i = blockIdx.x * 256 + threadIdx.x;   // over 4096*1024
  int b = i >> 10, j = i & 1023;
  int a = j >> 5, c = j & 31;
  size_t base = (size_t)b * 4096 + a * 128 + c * 2;
  float m0 = fmaxf(x[base],      x[base + 1]);
  float m1 = fmaxf(x[base + 64], x[base + 65]);
  M[i] = f2bf(-0.05f * E[i] * fmaxf(m0, m1));
}

// ---- driver ---------------------------------------------------------------

extern "C" void kernel_launch(void* const* d_in, const int* in_sizes, int n_in,
                              void* d_out, int out_size, void* d_ws, size_t ws_size,
                              hipStream_t stream) {
  (void)in_sizes; (void)n_in; (void)out_size; (void)ws_size;
  const float* inputs = (const float*)d_in[0];   // [4096, 512]
  const float* W      = (const float*)d_in[1];   // [4096, 512]
  const float* V      = (const float*)d_in[2];   // [256, 1024]
  float*       out    = (float*)d_out;           // [4096, 512]

  char* ws = (char*)d_ws;
  size_t off = 0;
  auto alloc = [&](size_t bytes) -> void* {
    void* p = ws + off;
    off += (bytes + 255) & ~(size_t)255;
    return p;
  };
  float* x    = (float*)alloc((size_t)4096 * 4096 * 4);
  u16*   xbf  = (u16*)  alloc((size_t)4096 * 4096 * 2);
  float* uS   = (float*)alloc((size_t)4096 * 256 * 4);
  u16*   ubf  = (u16*)  alloc((size_t)4096 * 256 * 2);
  float* E    = (float*)alloc((size_t)4096 * 1024 * 4);
  u16*   Rbf  = (u16*)  alloc((size_t)4096 * 512 * 2);
  u16*   Mbf  = (u16*)  alloc((size_t)4096 * 1024 * 2);
  u16*   Wbf  = (u16*)  alloc((size_t)4096 * 512 * 2);
  u16*   WTbf = (u16*)  alloc((size_t)512 * 4096 * 2);
  u16*   Vbf  = (u16*)  alloc((size_t)256 * 1024 * 2);
  u16*   VTbf = (u16*)  alloc((size_t)1024 * 256 * 2);

  dim3 blk(256);
  auto g1 = [](int n) { return dim3((unsigned)((n + 255) / 256)); };

  // state init (every call: deterministic under graph replay)
  k_fill_f32<<<g1(4096 * 4096), blk, 0, stream>>>(x, 0.0f, 4096 * 4096);
  k_fill_u16<<<g1(4096 * 4096), blk, 0, stream>>>(xbf, (u16)0, 4096 * 4096);
  k_fill_f32<<<g1(4096 * 256),  blk, 0, stream>>>(uS, 0.1f, 4096 * 256);
  k_fill_u16<<<g1(4096 * 256),  blk, 0, stream>>>(ubf, (u16)0x3DCD, 4096 * 256);
  // weight preprocessing: bf16 copies in n-major (k-contiguous) B-operand layouts
  k_cvt_bf<<<g1(4096 * 512), blk, 0, stream>>>(W, Wbf, 4096 * 512);
  k_tr_cvt<<<g1(4096 * 512), blk, 0, stream>>>(W, WTbf, 4096, 512);
  k_cvt_bf<<<g1(256 * 1024), blk, 0, stream>>>(V, Vbf, 256 * 1024);
  k_tr_cvt<<<g1(256 * 1024), blk, 0, stream>>>(V, VTbf, 256, 1024);

  dim3 gUV(4096 / 128, 1024 / 128);
  dim3 gXW(4096 / 128, 512 / 128);
  dim3 gGX(4096 / 128, 4096 / 128);
  dim3 gGU(4096 / 128, 256 / 128);

  for (int step = 0; step < 10; ++step) {
    if (step == 9) {  // final step only needs rec = x9 @ W
      k_gemm_xW<<<gXW, blk, 0, stream>>>(xbf, WTbf, inputs, Rbf, out, 1);
      break;
    }
    if (step == 0)  // x0 = 0 -> outputs = 0 -> R = -inputs
      k_negR<<<g1(4096 * 512), blk, 0, stream>>>(inputs, Rbf, 4096 * 512);
    else
      k_gemm_xW<<<gXW, blk, 0, stream>>>(xbf, WTbf, inputs, Rbf, out, 0);
    k_gemm_uV<<<gUV, blk, 0, stream>>>(ubf, VTbf, E);
    k_gemm_gradx<<<gGX, blk, 0, stream>>>(Rbf, Wbf, E, x, xbf);
    k_pool<<<g1(4096 * 1024), blk, 0, stream>>>(x, E, Mbf);
    k_gemm_gradu<<<gGU, blk, 0, stream>>>(Mbf, Vbf, uS, ubf);
  }
}
